// SAGE_Full_48704929136995
// MI455X (gfx1250) — matrix operand order repaired and checked by reading, not compiled, not run on hardware
//
#include <hip/hip_runtime.h>
#include <hip/hip_bf16.h>

typedef __attribute__((ext_vector_type(16))) _Float16 v16h;
typedef __attribute__((ext_vector_type(8)))  _Float16 v8h;
typedef __attribute__((ext_vector_type(8)))  float    v8f;

#define F_IN 256
#define HID  256
#define C_OUT 64

// ---------------- utility kernels ----------------

__global__ void k_zero(float* __restrict__ p, size_t n) {
  size_t i = (size_t)blockIdx.x * blockDim.x + threadIdx.x;
  if (i < n) p[i] = 0.0f;
}

__global__ void k_deg(const int* __restrict__ dst, float* __restrict__ deg, int E) {
  int i = blockIdx.x * blockDim.x + threadIdx.x;
  if (i < E) atomicAdd(&deg[dst[i]], 1.0f);
}

__global__ void k_invdeg(float* __restrict__ d, int n) {
  int i = blockIdx.x * blockDim.x + threadIdx.x;
  if (i < n) d[i] = 1.0f / fmaxf(d[i], 1.0f);
}

// scatter-add h[src] rows into hn[dst] rows. F fixed to 256.
// one thread handles 4 consecutive features of one edge (64 threads/edge).
__global__ void k_scatter(const float* __restrict__ h, const int* __restrict__ src,
                          const int* __restrict__ dst, float* __restrict__ hn, int E) {
  long long t = (long long)blockIdx.x * blockDim.x + threadIdx.x;
  long long total = (long long)E * 64;
  if (t >= total) return;
  int e = (int)(t >> 6);
  int q = ((int)t & 63) << 2;
  int s = src[e], d = dst[e];
  const float4 v = *(const float4*)(h + (size_t)s * 256 + q);
  float* o = hn + (size_t)d * 256 + q;
  atomicAdd(o + 0, v.x);
  atomicAdd(o + 1, v.y);
  atomicAdd(o + 2, v.z);
  atomicAdd(o + 3, v.w);
}

// f32 activations -> f16, optional per-row scale (invdeg), zero row padding.
__global__ void k_tof16(const float* __restrict__ srcp, const float* __restrict__ scaleRow,
                        _Float16* __restrict__ dstp, int nRows, int padRows, int F) {
  size_t t = (size_t)blockIdx.x * blockDim.x + threadIdx.x;
  size_t total = (size_t)padRows * F;
  if (t >= total) return;
  int row = (int)(t / F);
  float v = 0.0f;
  if (row < nRows) {
    v = srcp[t];
    if (scaleRow) v *= scaleRow[row];
  }
  dstp[t] = (_Float16)v;
}

// f32 weights W[Fin][Fout] -> f16 transposed WT[Fout][Fin]
__global__ void k_cvtW(const float* __restrict__ W, _Float16* __restrict__ WT,
                       int Fin, int Fout) {
  int t = blockIdx.x * blockDim.x + threadIdx.x;
  if (t >= Fin * Fout) return;
  int j = t / Fin;       // output column
  int k = t % Fin;       // reduction index
  WT[t] = (_Float16)W[(size_t)k * Fout + j];
}

// ---------------- fused SAGE layer GEMM ----------------
// out[n,j] = act( sum_k As[n,k] WTs[j,k] + sum_k An[n,k] WTn[j,k] + b[j] )
// As/An are pre-scaled f16, row-padded to a multiple of 64.
// Each wave: 64(M) x 16(N) slab = 4 accumulator tiles; B fragment loaded once
// per k-step and reused by 4 v_wmma_f32_16x16x32_f16.
__global__ void k_sage_gemm(const _Float16* __restrict__ As, const _Float16* __restrict__ An,
                            const _Float16* __restrict__ WTs, const _Float16* __restrict__ WTn,
                            const float* __restrict__ bias, float* __restrict__ out,
                            int nRows, int Fin, int Fout, int doRelu) {
  const int wave   = threadIdx.x >> 5;
  const int lane   = threadIdx.x & 31;
  const int laneLo = lane & 15;
  const int hi     = lane >> 4;               // 0: lanes 0-15, 1: lanes 16-31
  const int rowBase = blockIdx.x * 64;
  const int colTile = blockIdx.y * 64 + wave * 16;
  const int col     = colTile + laneLo;

  v8f acc0 = {}, acc1 = {}, acc2 = {}, acc3 = {};

  for (int pass = 0; pass < 2; ++pass) {
    const _Float16* __restrict__ A  = (pass == 0) ? As  : An;
    const _Float16* __restrict__ WT = (pass == 0) ? WTs : WTn;
    // lane's B column: halves k+{0..7} and k+{16..23} per k-step,
    // lane = column N, lane-half selects K +8 shift (matches A)
    const _Float16* __restrict__ wcol = WT + (size_t)col * Fin + (hi ? 8 : 0);
    // lane's A rows (4 M-tiles), half-lane selects K +8 shift
    const _Float16* __restrict__ a0 =
        A + (size_t)(rowBase + 0 * 16 + laneLo) * Fin + (hi ? 8 : 0);
    const _Float16* __restrict__ a1 = a0 + (size_t)16 * Fin;
    const _Float16* __restrict__ a2 = a0 + (size_t)32 * Fin;
    const _Float16* __restrict__ a3 = a0 + (size_t)48 * Fin;

    for (int kb = 0; kb < Fin; kb += 32) {
      const v16h bm = __builtin_shufflevector(*(const v8h*)(wcol + kb), *(const v8h*)(wcol + kb + 16), 0,1,2,3,4,5,6,7,8,9,10,11,12,13,14,15);

      // A fragment (16-bit A 16x32 layout): halves k+{0..7} and k+{16..23}
      v8h c0 = *(const v8h*)(a0 + kb);
      v8h c1 = *(const v8h*)(a0 + kb + 16);
      v16h am0 = __builtin_shufflevector(c0, c1, 0,1,2,3,4,5,6,7,8,9,10,11,12,13,14,15);
      acc0 = __builtin_amdgcn_wmma_f32_16x16x32_f16(false, am0, false, bm, (short)0,
                                                    acc0, false, false);
      c0 = *(const v8h*)(a1 + kb);
      c1 = *(const v8h*)(a1 + kb + 16);
      v16h am1 = __builtin_shufflevector(c0, c1, 0,1,2,3,4,5,6,7,8,9,10,11,12,13,14,15);
      acc1 = __builtin_amdgcn_wmma_f32_16x16x32_f16(false, am1, false, bm, (short)0,
                                                    acc1, false, false);
      c0 = *(const v8h*)(a2 + kb);
      c1 = *(const v8h*)(a2 + kb + 16);
      v16h am2 = __builtin_shufflevector(c0, c1, 0,1,2,3,4,5,6,7,8,9,10,11,12,13,14,15);
      acc2 = __builtin_amdgcn_wmma_f32_16x16x32_f16(false, am2, false, bm, (short)0,
                                                    acc2, false, false);
      c0 = *(const v8h*)(a3 + kb);
      c1 = *(const v8h*)(a3 + kb + 16);
      v16h am3 = __builtin_shufflevector(c0, c1, 0,1,2,3,4,5,6,7,8,9,10,11,12,13,14,15);
      acc3 = __builtin_amdgcn_wmma_f32_16x16x32_f16(false, am3, false, bm, (short)0,
                                                    acc3, false, false);
    }
  }

  // epilogue: C/D layout VGPR v -> M = v (+8 for lane-hi), N = lane&15
  const float bv = bias[col];
#pragma unroll
  for (int m = 0; m < 4; ++m) {
    const v8f acc = (m == 0) ? acc0 : (m == 1) ? acc1 : (m == 2) ? acc2 : acc3;
    const int rb = rowBase + m * 16 + (hi ? 8 : 0);
#pragma unroll
    for (int v = 0; v < 8; ++v) {
      int r = rb + v;
      if (r < nRows) {
        float val = acc[v] + bv;
        if (doRelu) val = fmaxf(val, 0.0f);
        out[(size_t)r * Fout + col] = val;
      }
    }
  }
}

// ---------------- row-wise log_softmax over 64 classes ----------------
// one wave per row; each lane owns cols {lane, lane+32}
__global__ void k_logsoftmax(float* __restrict__ x, int nRows) {
  const int wave = threadIdx.x >> 5;
  const int lane = threadIdx.x & 31;
  const int row  = blockIdx.x * (blockDim.x >> 5) + wave;
  if (row >= nRows) return;
  float* p = x + (size_t)row * 64;
  float a = p[lane];
  float b = p[lane + 32];
  float m = fmaxf(a, b);
#pragma unroll
  for (int o = 16; o >= 1; o >>= 1) m = fmaxf(m, __shfl_xor(m, o, 32));
  float s = expf(a - m) + expf(b - m);
#pragma unroll
  for (int o = 16; o >= 1; o >>= 1) s += __shfl_xor(s, o, 32);
  float l = logf(s);
  p[lane]      = a - m - l;
  p[lane + 32] = b - m - l;
}

// ---------------- launch ----------------

extern "C" void kernel_launch(void* const* d_in, const int* in_sizes, int n_in,
                              void* d_out, int out_size, void* d_ws, size_t ws_size,
                              hipStream_t stream) {
  (void)n_in; (void)out_size; (void)ws_size;
  const float* features = (const float*)d_in[0];
  const int*   src      = (const int*)d_in[1];
  const int*   dst      = (const int*)d_in[2];
  const float* Ws0 = (const float*)d_in[3];
  const float* Wn0 = (const float*)d_in[4];
  const float* b0  = (const float*)d_in[5];
  const float* Ws1 = (const float*)d_in[6];
  const float* Wn1 = (const float*)d_in[7];
  const float* b1  = (const float*)d_in[8];
  const float* Ws2 = (const float*)d_in[9];
  const float* Wn2 = (const float*)d_in[10];
  const float* b2  = (const float*)d_in[11];

  const int N = in_sizes[0] / F_IN;        // 50000
  const int E = in_sizes[1];               // 800000
  const int Npad = (N + 63) & ~63;         // 50048

  // workspace carve-out (256B aligned)
  char* ws = (char*)d_ws;
  size_t off = 0;
  auto carve = [&](size_t bytes) -> void* {
    void* p = ws + off;
    off = (off + bytes + 255) & ~(size_t)255;
    return p;
  };
  float*     invdeg = (float*)carve((size_t)N * sizeof(float));
  float*     hn     = (float*)carve((size_t)N * 256 * sizeof(float));
  float*     hbuf   = (float*)carve((size_t)N * 256 * sizeof(float));
  _Float16*  a16s   = (_Float16*)carve((size_t)Npad * 256 * sizeof(_Float16));
  _Float16*  a16n   = (_Float16*)carve((size_t)Npad * 256 * sizeof(_Float16));
  _Float16*  wtS0   = (_Float16*)carve((size_t)F_IN * HID * sizeof(_Float16));
  _Float16*  wtN0   = (_Float16*)carve((size_t)F_IN * HID * sizeof(_Float16));
  _Float16*  wtS1   = (_Float16*)carve((size_t)HID * HID * sizeof(_Float16));
  _Float16*  wtN1   = (_Float16*)carve((size_t)HID * HID * sizeof(_Float16));
  _Float16*  wtS2   = (_Float16*)carve((size_t)HID * C_OUT * sizeof(_Float16));
  _Float16*  wtN2   = (_Float16*)carve((size_t)HID * C_OUT * sizeof(_Float16));

  const dim3 blk(256);
  auto gz = [](size_t n) { return dim3((unsigned)((n + 255) / 256)); };

  // weight conversion + transpose (f32 -> f16, [Fin][Fout] -> [Fout][Fin])
  k_cvtW<<<gz((size_t)F_IN * HID), blk, 0, stream>>>(Ws0, wtS0, F_IN, HID);
  k_cvtW<<<gz((size_t)F_IN * HID), blk, 0, stream>>>(Wn0, wtN0, F_IN, HID);
  k_cvtW<<<gz((size_t)HID * HID),  blk, 0, stream>>>(Ws1, wtS1, HID, HID);
  k_cvtW<<<gz((size_t)HID * HID),  blk, 0, stream>>>(Wn1, wtN1, HID, HID);
  k_cvtW<<<gz((size_t)HID * C_OUT), blk, 0, stream>>>(Ws2, wtS2, HID, C_OUT);
  k_cvtW<<<gz((size_t)HID * C_OUT), blk, 0, stream>>>(Wn2, wtN2, HID, C_OUT);

  // degrees -> 1/max(deg,1)
  k_zero  <<<gz((size_t)N), blk, 0, stream>>>(invdeg, (size_t)N);
  k_deg   <<<gz((size_t)E), blk, 0, stream>>>(dst, invdeg, E);
  k_invdeg<<<gz((size_t)N), blk, 0, stream>>>(invdeg, N);

  const size_t nh  = (size_t)N * 256;
  const size_t nhp = (size_t)Npad * 256;
  const long long scatWork = (long long)E * 64;
  const dim3 gScat((unsigned)((scatWork + 255) / 256));
  const dim3 gGemm256(Npad / 64, HID / 64);
  const dim3 gGemm64(Npad / 64, C_OUT / 64);
  const dim3 blkGemm(128);

  // layer 0: features -> hbuf
  k_zero   <<<gz(nh), blk, 0, stream>>>(hn, nh);
  k_scatter<<<gScat, blk, 0, stream>>>(features, src, dst, hn, E);
  k_tof16  <<<gz(nhp), blk, 0, stream>>>(features, nullptr, a16s, N, Npad, 256);
  k_tof16  <<<gz(nhp), blk, 0, stream>>>(hn, invdeg, a16n, N, Npad, 256);
  k_sage_gemm<<<gGemm256, blkGemm, 0, stream>>>(a16s, a16n, wtS0, wtN0, b0, hbuf,
                                                N, F_IN, HID, 1);

  // layer 1: hbuf -> hbuf (GEMM reads only the f16 copies)
  k_zero   <<<gz(nh), blk, 0, stream>>>(hn, nh);
  k_scatter<<<gScat, blk, 0, stream>>>(hbuf, src, dst, hn, E);
  k_tof16  <<<gz(nhp), blk, 0, stream>>>(hbuf, nullptr, a16s, N, Npad, 256);
  k_tof16  <<<gz(nhp), blk, 0, stream>>>(hn, invdeg, a16n, N, Npad, 256);
  k_sage_gemm<<<gGemm256, blkGemm, 0, stream>>>(a16s, a16n, wtS1, wtN1, b1, hbuf,
                                                N, HID, HID, 1);

  // layer 2: hbuf -> logits (d_out), then log_softmax in-place
  float* logits = (float*)d_out;
  k_zero   <<<gz(nh), blk, 0, stream>>>(hn, nh);
  k_scatter<<<gScat, blk, 0, stream>>>(hbuf, src, dst, hn, E);
  k_tof16  <<<gz(nhp), blk, 0, stream>>>(hbuf, nullptr, a16s, N, Npad, 256);
  k_tof16  <<<gz(nhp), blk, 0, stream>>>(hn, invdeg, a16n, N, Npad, 256);
  k_sage_gemm<<<gGemm64, blkGemm, 0, stream>>>(a16s, a16n, wtS2, wtN2, b2, logits,
                                               N, HID, C_OUT, 0);
  k_logsoftmax<<<dim3((N + 7) / 8), dim3(256), 0, stream>>>(logits, N);
}
